// PointNetpp_19207093748189
// MI455X (gfx1250) — compile-verified, matrix operand order checked by reading, not run on hardware
//
#include <hip/hip_runtime.h>

// ---------------------------------------------------------------------------
// PointNet++ forward for MI455X (gfx1250, wave32).
// All dense layers run on v_wmma_f32_16x16x32_f16 (f16 in, f32 accumulate).
// ---------------------------------------------------------------------------

typedef _Float16 f16;
typedef __attribute__((ext_vector_type(2)))  _Float16 h2;
typedef __attribute__((ext_vector_type(8)))  _Float16 h8;
typedef __attribute__((ext_vector_type(16))) _Float16 v16h;
typedef __attribute__((ext_vector_type(8)))  float    v8f;

#define NPTS   16384
#define M1     4096
#define M2     1024
#define KNBR   64
#define SWLD   40          // padded LDS row stride (halves): 80B, 16B-aligned,
                           // bank starts 20*l mod 64 -> conflict-free b128 reads

// ---------------------------------------------------------------------------
// Generic GEMM:  O[M,Dout] = act(A[M,Kp](f16) @ W[Kp,Dout](f16) + bias)
// 256 threads = 8 waves; wave w -> rows [bx*128 + w*16, +16),
// cols [by*32, +32) as two 16x16 WMMA tiles sharing one A fragment.
// W tile (32K x 32N) staged transposed in LDS so each lane's B fragment is
// two contiguous 16B chunks (ds_load_b128 x2).
// A fragment layout (16-bit A 16x32, ISA 7.12.2): lane<16 holds K 0..7 &
// 16..23 of row (lane&15); lane>=16 holds K 8..15 & 24..31.  B symmetric.
// C/D: VGPR r -> row m0 + r + (lane>>4)*8, col n0 + (lane&15).
// ---------------------------------------------------------------------------
__global__ void __launch_bounds__(256)
k_gemm(const f16* __restrict__ A, const f16* __restrict__ W,
       const float* __restrict__ bias,
       f16* __restrict__ Oh, float* __restrict__ Of,
       int M, int Kp, int Dout, int do_relu)
{
  __shared__ f16 sW[32 * SWLD];              // [n][k] column-major, padded
  const int tid  = threadIdx.x;
  const int wave = tid >> 5;
  const int lane = tid & 31;
  const int m0   = blockIdx.x * 128 + wave * 16;
  const int n0   = blockIdx.y * 32;
  const int hi   = lane >> 4;                // 0 or 1
  const int l15  = lane & 15;
  const int kb   = hi * 8;

  v8f acc0 = {}, acc1 = {};
  const int arow = m0 + l15;

  for (int k0 = 0; k0 < Kp; k0 += 32) {
    // ---- stage 32(K) x 32(N) W tile, transposed, f16x2 global loads
#pragma unroll
    for (int it = 0; it < 2; ++it) {
      const int idx = tid + it * 256;        // 0..511
      const int kk  = idx >> 4;              // 0..31
      const int nn2 = (idx & 15) * 2;        // 0,2,...,30
      const int gn  = n0 + nn2;
      f16 w0 = (f16)0.f, w1 = (f16)0.f;
      if (gn + 1 < Dout) {                   // Dout even -> hot path
        const h2 w = *(const h2*)(W + (size_t)(k0 + kk) * Dout + gn);
        w0 = w[0]; w1 = w[1];
      } else if (gn < Dout) {
        w0 = W[(size_t)(k0 + kk) * Dout + gn];
      }
      sW[nn2 * SWLD + kk]       = w0;
      sW[(nn2 + 1) * SWLD + kk] = w1;
    }
    __syncthreads();

    // ---- A fragment: two contiguous 16B chunks (global_load_b128 x2)
    const h8* Ap = (const h8*)(A + (size_t)arow * Kp + k0 + kb);
    const h8 alo = Ap[0];                    // K kb..kb+7
    const h8 ahi = Ap[2];                    // K kb+16..kb+23
    const v16h a = __builtin_shufflevector(alo, ahi,
        0, 1, 2, 3, 4, 5, 6, 7, 8, 9, 10, 11, 12, 13, 14, 15);

    // ---- B fragments: contiguous per-lane chunks from LDS (ds_load_b128)
    const h8* B0 = (const h8*)(sW + (size_t)l15 * SWLD + kb);
    const h8* B1 = (const h8*)(sW + (size_t)(16 + l15) * SWLD + kb);
    const v16h b0 = __builtin_shufflevector(B0[0], B0[2],
        0, 1, 2, 3, 4, 5, 6, 7, 8, 9, 10, 11, 12, 13, 14, 15);
    const v16h b1 = __builtin_shufflevector(B1[0], B1[2],
        0, 1, 2, 3, 4, 5, 6, 7, 8, 9, 10, 11, 12, 13, 14, 15);

    acc0 = __builtin_amdgcn_wmma_f32_16x16x32_f16(
        false, a, false, b0, (short)0, acc0, false, false);
    acc1 = __builtin_amdgcn_wmma_f32_16x16x32_f16(
        false, a, false, b1, (short)0, acc1, false, false);
    __syncthreads();
  }

  // ---- epilogue: bias + optional ReLU, two 16-column tiles
#pragma unroll
  for (int j = 0; j < 2; ++j) {
    const int col = n0 + j * 16 + l15;
    if (col < Dout) {
      const float bv = bias[col];
      const v8f acc = j ? acc1 : acc0;
#pragma unroll
      for (int r = 0; r < 8; ++r) {
        const int row = m0 + r + hi * 8;
        float v = acc[r] + bv;
        if (do_relu) v = fmaxf(v, 0.f);
        if (Of) Of[(size_t)row * Dout + col] = v;
        else    Oh[(size_t)row * Dout + col] = (f16)v;
      }
    }
  }
}

// ---------- weight conversion: f32 [din x dout] -> f16 [kp x dout], zero pad
__global__ void k_cvt_w(const float* __restrict__ W, int din, int dout, int kp,
                        f16* __restrict__ Wh)
{
  const int t = blockIdx.x * 256 + threadIdx.x;
  if (t >= kp * dout) return;
  const int k = t / dout, n = t % dout;
  Wh[t] = (k < din) ? (f16)W[(size_t)k * dout + n] : (f16)0.f;
}

// ---------- x (N x 7, f32) -> xh (N x 32, f16 zero-padded)
__global__ void k_cvt_x(const float* __restrict__ x, f16* __restrict__ xh)
{
  const int t = blockIdx.x * 256 + threadIdx.x;   // N*32 threads
  const int row = t >> 5, col = t & 31;
  xh[t] = (col < 7) ? (f16)x[row * 7 + col] : (f16)0.f;
}

// ---------- pos0 = x[:, :2]
__global__ void k_pos_from_x(const float* __restrict__ x, float* __restrict__ p)
{
  const int t = blockIdx.x * 256 + threadIdx.x;
  p[2 * t]     = x[t * 7];
  p[2 * t + 1] = x[t * 7 + 1];
}

// ---------- pos_dst[i] = pos_src[idx[i]]
__global__ void k_gather_pos(const float* __restrict__ src,
                             const int* __restrict__ idx,
                             float* __restrict__ dst)
{
  const int t = blockIdx.x * 256 + threadIdx.x;
  const int s = idx[t];
  dst[2 * t]     = src[2 * s];
  dst[2 * t + 1] = src[2 * s + 1];
}

// ---------------------------------------------------------------------------
// Farthest point sampling (exact reference semantics): single 1024-thread
// block, min-distances in registers, tie-stable block argmax.
// ---------------------------------------------------------------------------
__global__ void __launch_bounds__(1024)
k_fps(const float* __restrict__ pos, int n, int nsample, int* __restrict__ idx_out)
{
  __shared__ float sv[1024];
  __shared__ int   si[1024];
  const int tid = threadIdx.x;
  const int chunks = (n + 1023) >> 10;

  float dl[16];
#pragma unroll
  for (int j = 0; j < 16; ++j) {
    const int p = tid + (j << 10);
    dl[j] = (j < chunks && p < n) ? 3.4e38f : -1.0f;
  }

  int cur = 0;
  for (int s = 0; s < nsample; ++s) {
    if (tid == 0) idx_out[s] = cur;
    const float cx = pos[2 * cur], cy = pos[2 * cur + 1];
    float bv = -2.0f; int bi = 0;
    for (int j = 0; j < chunks; ++j) {
      const int p = tid + (j << 10);
      if (p < n) {
        const float dx = pos[2 * p] - cx, dy = pos[2 * p + 1] - cy;
        const float v = fminf(dl[j], dx * dx + dy * dy);
        dl[j] = v;
        if (v > bv) { bv = v; bi = p; }          // ascending p -> stable
      }
    }
    sv[tid] = bv; si[tid] = bi;
    __syncthreads();
    for (int off = 512; off > 0; off >>= 1) {
      if (tid < off) {
        const float ov = sv[tid + off]; const int oi = si[tid + off];
        if (ov > sv[tid] || (ov == sv[tid] && oi < si[tid])) {
          sv[tid] = ov; si[tid] = oi;
        }
      }
      __syncthreads();
    }
    cur = si[0];
    __syncthreads();
  }
}

// ---------------------------------------------------------------------------
// Radius-bounded top-64 neighbors per center. Candidates (expected ~128)
// collected in LDS, bitonic-sorted on packed (d2_bits, idx) -> stable top-k.
// ---------------------------------------------------------------------------
__global__ void __launch_bounds__(256)
k_radius64(const float* __restrict__ pos, int nsrc,
           const int* __restrict__ centers, float r2,
           int* __restrict__ nbr, int* __restrict__ valid)
{
  __shared__ unsigned long long keys[1024];
  __shared__ int cnt;
  const int tid = threadIdx.x;
  const int c   = blockIdx.x;
  if (tid == 0) cnt = 0;
  __syncthreads();

  const int   ci = centers[c];
  const float cx = pos[2 * ci], cy = pos[2 * ci + 1];
  for (int i = tid; i < nsrc; i += 256) {
    const float dx = pos[2 * i] - cx, dy = pos[2 * i + 1] - cy;
    const float d2 = dx * dx + dy * dy;
    if (d2 <= r2) {
      const int p = atomicAdd(&cnt, 1);
      if (p < 1024)
        keys[p] = ((unsigned long long)__float_as_uint(d2) << 32) | (unsigned)i;
    }
  }
  __syncthreads();
  int m = cnt; if (m > 1024) m = 1024;
  for (int i = tid; i < 1024; i += 256)
    if (i >= m) keys[i] = ~0ull;
  __syncthreads();

  for (int k = 2; k <= 1024; k <<= 1)
    for (int j = k >> 1; j > 0; j >>= 1) {
      for (int i = tid; i < 1024; i += 256) {
        const int ixj = i ^ j;
        if (ixj > i) {
          const unsigned long long a = keys[i], b = keys[ixj];
          const bool up = (i & k) == 0;
          if ((a > b) == up) { keys[i] = b; keys[ixj] = a; }
        }
      }
      __syncthreads();
    }

  if (tid < KNBR) {
    const unsigned long long kk = keys[tid];
    bool ok = (kk != ~0ull);
    int  id = ok ? (int)(kk & 0xffffffffu) : ci;
    if (tid == 0) { id = ci; ok = true; }        // nbr[:,0] = center
    nbr[c * KNBR + tid]   = id;
    valid[c * KNBR + tid] = ok ? 1 : 0;
  }
}

// ---------- gathered local-conv input: [feat(nbr) | rel | 0-pad] as f16
__global__ void k_gather_local(const f16* __restrict__ feat, int fdim,
                               const float* __restrict__ pos_src,
                               const int* __restrict__ centers,
                               const int* __restrict__ nbr,
                               int outld, f16* __restrict__ out)
{
  const int row = blockIdx.x;                    // center*64 + k
  const int c   = row >> 6;
  const int col = threadIdx.x;                   // outld threads
  const int nb  = nbr[row];
  f16 v = (f16)0.f;
  if (col < fdim)           v = feat[(size_t)nb * fdim + col];
  else if (col == fdim)     v = (f16)(pos_src[2 * nb]     - pos_src[2 * centers[c]]);
  else if (col == fdim + 1) v = (f16)(pos_src[2 * nb + 1] - pos_src[2 * centers[c] + 1]);
  out[(size_t)row * outld + col] = v;
}

// ---------- masked max over the 64 neighbors (slot 0 always valid)
__global__ void k_maxpool64(const f16* __restrict__ L, const int* __restrict__ valid,
                            int dim, f16* __restrict__ out)
{
  const int c   = blockIdx.x;
  const int col = threadIdx.x;                   // dim threads
  const size_t base = (size_t)c * KNBR;
  float m = -3.4e38f;
  for (int k = 0; k < KNBR; ++k)
    if (valid[base + k]) m = fmaxf(m, (float)L[(base + k) * dim + col]);
  out[(size_t)c * dim + col] = (f16)m;
}

// ---------- 3-NN inverse-distance interpolation (reference semantics)
__global__ void k_knn3(const float* __restrict__ posf,
                       const float* __restrict__ posc, int nc,
                       const f16* __restrict__ xc, int dim,
                       f16* __restrict__ out)
{
  const int i = blockIdx.x * blockDim.x + threadIdx.x;
  const float fx = posf[2 * i], fy = posf[2 * i + 1];
  float d0 = 3.4e38f, d1 = 3.4e38f, d2 = 3.4e38f;
  int   i0 = 0, i1 = 0, i2 = 0;
  for (int j = 0; j < nc; ++j) {
    const float dx = fx - posc[2 * j], dy = fy - posc[2 * j + 1];
    const float d = dx * dx + dy * dy;
    if (d < d0)      { d2 = d1; i2 = i1; d1 = d0; i1 = i0; d0 = d; i0 = j; }
    else if (d < d1) { d2 = d1; i2 = i1; d1 = d; i1 = j; }
    else if (d < d2) { d2 = d; i2 = j; }
  }
  const float w0 = 1.f / fmaxf(d0, 1e-16f);
  const float w1 = 1.f / fmaxf(d1, 1e-16f);
  const float w2 = 1.f / fmaxf(d2, 1e-16f);
  const float ws = w0 + w1 + w2;
  const f16* x0 = xc + (size_t)i0 * dim;
  const f16* x1 = xc + (size_t)i1 * dim;
  const f16* x2 = xc + (size_t)i2 * dim;
  for (int c = 0; c < dim; ++c)
    out[(size_t)i * dim + c] =
        (f16)(((float)x0[c] * w0 + (float)x1[c] * w1 + (float)x2[c] * w2) / ws);
}

// ---------- dst[row] = [a[row] | b[row]]
__global__ void k_concat(const f16* __restrict__ a, int da,
                         const f16* __restrict__ b, int db,
                         f16* __restrict__ dst)
{
  const int row = blockIdx.x;
  const int c   = threadIdx.x;                   // da+db threads
  const int ld  = da + db;
  dst[(size_t)row * ld + c] =
      (c < da) ? a[(size_t)row * da + c] : b[(size_t)row * db + (c - da)];
}

// ---------------------------------------------------------------------------
// Host orchestration
// ---------------------------------------------------------------------------
static void gemm(hipStream_t s, const void* A, const void* W, const void* bias,
                 void* Oh, float* Of, int M, int Kp, int Dout, int relu)
{
  dim3 g(M / 128, (Dout + 31) / 32);
  k_gemm<<<g, 256, 0, s>>>((const f16*)A, (const f16*)W, (const float*)bias,
                           (f16*)Oh, Of, M, Kp, Dout, relu);
}

extern "C" void kernel_launch(void* const* d_in, const int* in_sizes, int n_in,
                              void* d_out, int out_size, void* d_ws, size_t ws_size,
                              hipStream_t stream)
{
  (void)in_sizes; (void)n_in; (void)out_size; (void)ws_size;
  const float* x = (const float*)d_in[0];

  char* base = (char*)d_ws;
  size_t off = 0;
  auto alloc = [&](size_t bytes) -> void* {
    off = (off + 255) & ~(size_t)255;
    void* p = base + off;
    off += bytes;
    return p;
  };

  // -------- persistent buffers
  float* pos0 = (float*)alloc((size_t)NPTS * 2 * 4);
  float* pos1 = (float*)alloc((size_t)M1 * 2 * 4);
  float* pos2 = (float*)alloc((size_t)M2 * 2 * 4);
  int* idx0 = (int*)alloc((size_t)M1 * 4);
  int* idx1 = (int*)alloc((size_t)M2 * 4);
  int* nbr0 = (int*)alloc((size_t)M1 * KNBR * 4);
  int* val0 = (int*)alloc((size_t)M1 * KNBR * 4);
  int* nbr1 = (int*)alloc((size_t)M2 * KNBR * 4);
  int* val1 = (int*)alloc((size_t)M2 * KNBR * 4);
  f16* xh    = (f16*)alloc((size_t)NPTS * 32 * 2);
  f16* z0h   = (f16*)alloc((size_t)NPTS * 64 * 2);
  f16* z0    = (f16*)alloc((size_t)NPTS * 64 * 2);
  f16* z1pre = (f16*)alloc((size_t)M1 * 128 * 2);
  f16* g0h   = (f16*)alloc((size_t)M1 * 128 * 2);
  f16* z1    = (f16*)alloc((size_t)M1 * 128 * 2);
  f16* z2pre = (f16*)alloc((size_t)M2 * 256 * 2);
  f16* g1h   = (f16*)alloc((size_t)M2 * 256 * 2);
  f16* z2    = (f16*)alloc((size_t)M2 * 256 * 2);

  // -------- f16 weights (K zero-padded to multiple of 32)
  f16* wE0  = (f16*)alloc(32 * 64 * 2);   f16* wE1  = (f16*)alloc(64 * 64 * 2);
  f16* wL00 = (f16*)alloc(96 * 64 * 2);   f16* wL01 = (f16*)alloc(64 * 128 * 2);
  f16* wG00 = (f16*)alloc(128 * 128 * 2); f16* wG01 = (f16*)alloc(128 * 128 * 2);
  f16* wL10 = (f16*)alloc(160 * 256 * 2); f16* wL11 = (f16*)alloc(256 * 256 * 2);
  f16* wG10 = (f16*)alloc(256 * 256 * 2); f16* wG11 = (f16*)alloc(256 * 256 * 2);
  f16* wF10 = (f16*)alloc(384 * 256 * 2); f16* wF11 = (f16*)alloc(256 * 128 * 2);
  f16* wF00 = (f16*)alloc(192 * 128 * 2); f16* wF01 = (f16*)alloc(128 * 64 * 2);
  f16* wD0  = (f16*)alloc(64 * 64 * 2);   f16* wD1  = (f16*)alloc(64 * 4 * 2);

  // -------- big ping-pong arenas for the per-neighbor tensors
  char* S1 = (char*)alloc(64ull << 20);   // A0(48M) -> L0o(64M) -> L1h(32M)
  char* S2 = (char*)alloc(32ull << 20);   // L0h(32M) -> A1(20M) -> L1o(32M)

  // Phase-2 buffers carved from the arenas (local-conv stage retired by then).
  f16* interp1 = (f16*)(S2);
  f16* cat1    = (f16*)(S2 + (4ull  << 20));
  f16* f1h     = (f16*)(S2 + (8ull  << 20));
  f16* hfp1    = (f16*)(S2 + (12ull << 20));
  f16* interp0 = (f16*)(S1);
  f16* cat0    = (f16*)(S1 + (8ull  << 20));
  f16* f0h     = (f16*)(S1 + (16ull << 20));
  f16* hfp0    = (f16*)(S1 + (24ull << 20));
  f16* dech    = (f16*)(S1 + (28ull << 20));

  // -------- input prep
  k_cvt_x<<<(NPTS * 32) / 256, 256, 0, stream>>>(x, xh);
  k_pos_from_x<<<NPTS / 256, 256, 0, stream>>>(x, pos0);

  auto cvtw = [&](int widx, int din, int dout, int kp, f16* dst) {
    const int total = kp * dout;
    k_cvt_w<<<(total + 255) / 256, 256, 0, stream>>>(
        (const float*)d_in[widx], din, dout, kp, dst);
  };
  cvtw(1,   7,  64,  32, wE0);  cvtw(3,  64,  64,  64, wE1);
  cvtw(5,  66,  64,  96, wL00); cvtw(7,  64, 128,  64, wL01);
  cvtw(9, 128, 128, 128, wG00); cvtw(11,128, 128, 128, wG01);
  cvtw(13,130, 256, 160, wL10); cvtw(15,256, 256, 256, wL11);
  cvtw(17,256, 256, 256, wG10); cvtw(19,256, 256, 256, wG11);
  cvtw(21,384, 256, 384, wF10); cvtw(23,256, 128, 256, wF11);
  cvtw(25,192, 128, 192, wF00); cvtw(27,128,  64, 128, wF01);
  cvtw(29, 64,  64,  64, wD0);  cvtw(31, 64,   4,  64, wD1);

  // -------- encoder
  gemm(stream, xh,  wE0, d_in[2], z0h, nullptr, NPTS, 32, 64, 1);
  gemm(stream, z0h, wE1, d_in[4], z0,  nullptr, NPTS, 64, 64, 1);

  // -------- level 0 -> 1: FPS, radius top-64, local conv, global MLP
  k_fps<<<1, 1024, 0, stream>>>(pos0, NPTS, M1, idx0);
  k_gather_pos<<<M1 / 256, 256, 0, stream>>>(pos0, idx0, pos1);
  k_radius64<<<M1, 256, 0, stream>>>(pos0, NPTS, idx0, 0.05f * 0.05f, nbr0, val0);
  k_gather_local<<<M1 * KNBR, 96, 0, stream>>>(z0, 64, pos0, idx0, nbr0, 96, (f16*)S1);
  gemm(stream, S1, wL00, d_in[6], S2, nullptr, M1 * KNBR, 96, 64, 1);
  gemm(stream, S2, wL01, d_in[8], S1, nullptr, M1 * KNBR, 64, 128, 1);
  k_maxpool64<<<M1, 128, 0, stream>>>((const f16*)S1, val0, 128, z1pre);
  gemm(stream, z1pre, wG00, d_in[10], g0h, nullptr, M1, 128, 128, 1);
  gemm(stream, g0h,   wG01, d_in[12], z1,  nullptr, M1, 128, 128, 1);

  // -------- level 1 -> 2
  k_fps<<<1, 1024, 0, stream>>>(pos1, M1, M2, idx1);
  k_gather_pos<<<M2 / 256, 256, 0, stream>>>(pos1, idx1, pos2);
  k_radius64<<<M2, 256, 0, stream>>>(pos1, M1, idx1, 0.1f * 0.1f, nbr1, val1);
  k_gather_local<<<M2 * KNBR, 160, 0, stream>>>(z1, 128, pos1, idx1, nbr1, 160, (f16*)S2);
  gemm(stream, S2, wL10, d_in[14], S1, nullptr, M2 * KNBR, 160, 256, 1);
  gemm(stream, S1, wL11, d_in[16], S2, nullptr, M2 * KNBR, 256, 256, 1);
  k_maxpool64<<<M2, 256, 0, stream>>>((const f16*)S2, val1, 256, z2pre);
  gemm(stream, z2pre, wG10, d_in[18], g1h, nullptr, M2, 256, 256, 1);
  gemm(stream, g1h,   wG11, d_in[20], z2,  nullptr, M2, 256, 256, 1);

  // -------- FP branch back down: level 2 -> 1
  k_knn3<<<M1 / 256, 256, 0, stream>>>(pos1, pos2, M2, z2, 256, interp1);
  k_concat<<<M1, 384, 0, stream>>>(interp1, 256, z1, 128, cat1);
  gemm(stream, cat1, wF10, d_in[22], f1h,  nullptr, M1, 384, 256, 1);
  gemm(stream, f1h,  wF11, d_in[24], hfp1, nullptr, M1, 256, 128, 1);

  // -------- level 1 -> 0
  k_knn3<<<NPTS / 256, 256, 0, stream>>>(pos0, pos1, M1, hfp1, 128, interp0);
  k_concat<<<NPTS, 192, 0, stream>>>(interp0, 128, z0, 64, cat0);
  gemm(stream, cat0, wF00, d_in[26], f0h,  nullptr, NPTS, 192, 128, 1);
  gemm(stream, f0h,  wF01, d_in[28], hfp0, nullptr, NPTS, 128, 64, 1);

  // -------- decoder (last layer: no ReLU, f32 straight to d_out)
  gemm(stream, hfp0, wD0, d_in[30], dech, nullptr, NPTS, 64, 64, 1);
  gemm(stream, dech, wD1, d_in[32], nullptr, (float*)d_out, NPTS, 64, 4, 0);
}
